// QAGNN_6519760355933
// MI455X (gfx1250) — matrix operand order repaired and checked by reading, not compiled, not run on hardware
//
#include <hip/hip_runtime.h>
#include <hip/hip_bf16.h>

#define DIN 768
#define DH  256
#define EPS_BN 1e-5f
#define EPS_COS 1e-8f

typedef _Float16 v16h __attribute__((ext_vector_type(16)));
typedef _Float16 v8h  __attribute__((ext_vector_type(8)));
typedef float    v8f  __attribute__((ext_vector_type(8)));

// ---------------- order-preserving float<->int encoding for atomicMax ----------
__device__ __forceinline__ int enc_f(float f) {
    int i = __float_as_int(f);
    return (i >= 0) ? i : (i ^ 0x7fffffff);
}
__device__ __forceinline__ float dec_i(int i) {
    return __int_as_float((i >= 0) ? i : (i ^ 0x7fffffff));
}

// ---------------- 1. cosine-relevance gating: h0 = x * rel  (f32 -> f16) ------
__global__ void qa_gate_kernel(const float* __restrict__ ce, const float* __restrict__ x,
                               const int* __restrict__ batch, _Float16* __restrict__ h0,
                               int n_nodes) {
    int wid  = (blockIdx.x * blockDim.x + threadIdx.x) >> 5;
    int lane = threadIdx.x & 31;
    if (wid >= n_nodes) return;
    const float* xr = x  + (size_t)wid * DIN;
    const float* cr = ce + (size_t)batch[wid] * DIN;
    float dot = 0.f, nx = 0.f, nc = 0.f;
    for (int k = lane; k < DIN; k += 32) {
        float a = cr[k], b = xr[k];
        dot += a * b; nx += b * b; nc += a * a;
    }
    for (int off = 16; off; off >>= 1) {
        dot += __shfl_xor(dot, off);
        nx  += __shfl_xor(nx,  off);
        nc  += __shfl_xor(nc,  off);
    }
    float rel = dot / fmaxf(sqrtf(nc) * sqrtf(nx), EPS_COS);
    _Float16* hr = h0 + (size_t)wid * DIN;
    for (int k = lane; k < DIN; k += 32) hr[k] = (_Float16)(xr[k] * rel);
}

// ---------------- 2. W [K x N] f32 -> Wt [N x K] f16 --------------------------
__global__ void qa_wt_kernel(const float* __restrict__ W, _Float16* __restrict__ Wt,
                             int K, int N) {
    int idx = blockIdx.x * blockDim.x + threadIdx.x;
    if (idx >= K * N) return;
    int n = idx / K, k = idx - n * K;
    Wt[(size_t)n * K + k] = (_Float16)W[(size_t)k * N + n];
}

// ---------------- 3. WMMA GEMM: C[MxN] = A[MxK] * Bt[NxK]^T  (f16 -> f32) -----
// one wave computes a 16x64 strip (4 wmma accumulators); 8 waves/block.
// K-loop is double-buffered: fragments for k+32 are issued before the 4 WMMAs
// of step k, so global loads overlap matrix math within a single wave.
__device__ __forceinline__ void qa_load_frags(const _Float16* __restrict__ arow,
                                              const _Float16* __restrict__ Bt,
                                              int K, int n0, int half16, int hiA, int hiB,
                                              int k0, v16h& af, v16h* bf) {
    union U { v16h v; v8h h[2]; };
    U ua;
    ua.h[0] = *(const v8h*)(arow + k0 + hiA);
    ua.h[1] = *(const v8h*)(arow + k0 + 16 + hiA);
    af = ua.v;
#pragma unroll
    for (int j = 0; j < 4; ++j) {
        const _Float16* brow = Bt + (size_t)(n0 + j * 16 + half16) * K + k0 + hiB;
        U ub;
        ub.h[0] = *(const v8h*)(brow);
        ub.h[1] = *(const v8h*)(brow + 16);
        bf[j] = ub.v;
    }
}

__global__ void qa_wmma_gemm_kernel(const _Float16* __restrict__ A,
                                    const _Float16* __restrict__ Bt,
                                    float* __restrict__ C,
                                    int M, int K, int N) {
    const int lane   = threadIdx.x & 31;
    const int wave   = threadIdx.x >> 5;
    const int tileM  = blockIdx.x * 8 + wave;
    if (tileM * 16 >= M) return;                     // wave-uniform: EXEC stays all-1s
    const int n0     = blockIdx.y * 64;
    const int half16 = lane & 15;
    const int hiA    = (lane >> 4) ? 8 : 0;          // A: K offset inside 16-group
    const int hiB    = (lane >> 4) ? 8 : 0;          // B: K offset inside 16-group
    const int rowAdd = (lane >> 4) ? 8 : 0;          // C/D: M offset

    int mr = tileM * 16 + half16;
    if (mr >= M) mr = M - 1;                         // clamp for ragged tail
    const _Float16* arow = A + (size_t)mr * K;

    v8f acc[4] = {v8f{}, v8f{}, v8f{}, v8f{}};

    v16h af, bf[4];
    qa_load_frags(arow, Bt, K, n0, half16, hiA, hiB, 0, af, bf);

    for (int k0 = 0; k0 < K; k0 += 32) {
        const int kn = (k0 + 32 < K) ? (k0 + 32) : 0; // clamped: extra loads on tail, unused
        if (k0 + 64 < K) __builtin_prefetch(arow + k0 + 64 + hiA, 0, 3);
        v16h afN, bfN[4];
        qa_load_frags(arow, Bt, K, n0, half16, hiA, hiB, kn, afN, bfN);
#pragma unroll
        for (int j = 0; j < 4; ++j)
            acc[j] = __builtin_amdgcn_wmma_f32_16x16x32_f16(
                false, af, false, bf[j], (short)0, acc[j], false, false);
        af = afN;
#pragma unroll
        for (int j = 0; j < 4; ++j) bf[j] = bfN[j];
    }
#pragma unroll
    for (int r = 0; r < 8; ++r) {
        int m = tileM * 16 + r + rowAdd;
        if (m >= M) continue;
        float* crow = C + (size_t)m * N + n0 + half16;
        crow[0]  = acc[0][r];
        crow[16] = acc[1][r];
        crow[32] = acc[2][r];
        crow[48] = acc[3][r];
    }
}

// ---------------- 4. per-node attention logit pieces --------------------------
__global__ void qa_alpha_kernel(const float* __restrict__ h, const float* __restrict__ a_s,
                                const float* __restrict__ a_d, float* __restrict__ als,
                                float* __restrict__ ald, int n_nodes) {
    int wid  = (blockIdx.x * blockDim.x + threadIdx.x) >> 5;
    int lane = threadIdx.x & 31;
    if (wid >= n_nodes) return;
    const float* hr = h + (size_t)wid * DH;
    float ss = 0.f, sd = 0.f;
    for (int k = lane; k < DH; k += 32) {
        float v = hr[k];
        ss += v * a_s[k];
        sd += v * a_d[k];
    }
    for (int off = 16; off; off >>= 1) {
        ss += __shfl_xor(ss, off);
        sd += __shfl_xor(sd, off);
    }
    if (lane == 0) { als[wid] = ss; ald[wid] = sd; }
}

// ---------------- 5. segmented softmax over edges (3 passes) ------------------
__global__ void qa_seg_init_kernel(int* __restrict__ mx, float* __restrict__ den, int n) {
    int i = blockIdx.x * blockDim.x + threadIdx.x;
    if (i < n) { mx[i] = 0x80000000; den[i] = 0.f; }
}

__global__ void qa_edge_max_kernel(const int* __restrict__ src, const int* __restrict__ dst,
                                   const float* __restrict__ als, const float* __restrict__ ald,
                                   float* __restrict__ elog, int* __restrict__ mx,
                                   int E, int n_nodes) {
    int e = blockIdx.x * blockDim.x + threadIdx.x;
    if (e >= E + n_nodes) return;
    int s, d;
    if (e < E) { s = src[e]; d = dst[e]; } else { s = d = e - E; }
    float v = als[s] + ald[d];
    float l = (v > 0.f) ? v : 0.2f * v;              // leaky_relu 0.2
    elog[e] = l;
    atomicMax(&mx[d], enc_f(l));
}

__global__ void qa_edge_exp_kernel(const int* __restrict__ src, const int* __restrict__ dst,
                                   const float* __restrict__ elog, const int* __restrict__ mx,
                                   float* __restrict__ wgt, float* __restrict__ den,
                                   int E, int n_nodes) {
    int e = blockIdx.x * blockDim.x + threadIdx.x;
    if (e >= E + n_nodes) return;
    int d = (e < E) ? dst[e] : (e - E);
    float wv = __expf(elog[e] - dec_i(mx[d]));
    wgt[e] = wv;
    atomicAdd(&den[d], wv);
}

__global__ void qa_out_bias_kernel(float* __restrict__ out, const float* __restrict__ b, int total) {
    int i = blockIdx.x * blockDim.x + threadIdx.x;
    if (i < total) out[i] = b[i & (DH - 1)];
}

// one block per edge, one column per thread: out[dst] += h[src] * (w/den[dst])
__global__ void qa_edge_agg_kernel(const int* __restrict__ src, const int* __restrict__ dst,
                                   const float* __restrict__ h, const float* __restrict__ wgt,
                                   const float* __restrict__ den, float* __restrict__ out,
                                   int E, int n_nodes) {
    int e = blockIdx.x;
    int c = threadIdx.x;
    int s, d;
    if (e < E) { s = src[e]; d = dst[e]; } else { s = d = e - E; }
    float coef = wgt[e] / den[d];
    atomicAdd(&out[(size_t)d * DH + c], h[(size_t)s * DH + c] * coef);
}

// ---------------- 6. batchnorm (population stats) + relu -> f16 A2 ------------
__global__ void qa_bn_stats_kernel(const float* __restrict__ h, float* __restrict__ stat,
                                   int n_nodes) {
    int c = blockIdx.x, t = threadIdx.x;
    float s = 0.f, s2 = 0.f;
    for (int i = t; i < n_nodes; i += blockDim.x) {
        float v = h[(size_t)i * DH + c];
        s += v; s2 += v * v;
    }
    __shared__ float sh[256], sh2[256];
    sh[t] = s; sh2[t] = s2; __syncthreads();
    for (int off = 128; off; off >>= 1) {
        if (t < off) { sh[t] += sh[t + off]; sh2[t] += sh2[t + off]; }
        __syncthreads();
    }
    if (t == 0) {
        float m = sh[0] / (float)n_nodes;
        stat[c]      = m;
        stat[DH + c] = sh2[0] / (float)n_nodes - m * m;
    }
}

__global__ void qa_bn_apply_kernel(const float* __restrict__ h, const float* __restrict__ stat,
                                   const float* __restrict__ gamma, const float* __restrict__ beta,
                                   _Float16* __restrict__ a2, int total) {
    int i = blockIdx.x * blockDim.x + threadIdx.x;
    if (i >= total) return;
    int c = i & (DH - 1);
    float v = gamma[c] * (h[i] - stat[c]) * rsqrtf(stat[DH + c] + EPS_BN) + beta[c];
    a2[i] = (_Float16)fmaxf(v, 0.f);
}

// ---------------- 7. pooling + final linear -----------------------------------
__global__ void qa_zero_kernel(float* __restrict__ p, int n) {
    int i = blockIdx.x * blockDim.x + threadIdx.x;
    if (i < n) p[i] = 0.f;
}

__global__ void qa_counts_kernel(const int* __restrict__ batch, float* __restrict__ cnts, int n) {
    int i = blockIdx.x * blockDim.x + threadIdx.x;
    if (i < n) atomicAdd(&cnts[batch[i]], 1.0f);
}

__global__ void qa_pool_kernel(const float* __restrict__ h, const int* __restrict__ batch,
                               float* __restrict__ pool, int total) {
    int i = blockIdx.x * blockDim.x + threadIdx.x;
    if (i >= total) return;
    int node = i >> 8;                               // DH == 256
    float v = fmaxf(h[i], 0.f);                      // final relu fused here
    atomicAdd(&pool[(size_t)batch[node] * DH + (i & (DH - 1))], v);
}

__global__ void qa_final_kernel(const float* __restrict__ pool, const float* __restrict__ cnts,
                                const float* __restrict__ ce, const float* __restrict__ Wc,
                                const float* __restrict__ bc, float* __restrict__ out) {
    int g = blockIdx.x, t = threadIdx.x;
    float cnt = fmaxf(cnts[g], 1.0f);
    float s = 0.f;
    for (int c = t; c < DH; c += blockDim.x)
        s += (pool[(size_t)g * DH + c] / cnt) * Wc[c];
    for (int k = t; k < DIN; k += blockDim.x)
        s += ce[(size_t)g * DIN + k] * Wc[DH + k];
    __shared__ float sh[256];
    sh[t] = s; __syncthreads();
    for (int off = 128; off; off >>= 1) {
        if (t < off) sh[t] += sh[t + off];
        __syncthreads();
    }
    if (t == 0) out[g] = sh[0] + bc[0];
}

// ---------------- launcher ----------------------------------------------------
extern "C" void kernel_launch(void* const* d_in, const int* in_sizes, int n_in,
                              void* d_out, int out_size, void* d_ws, size_t ws_size,
                              hipStream_t stream) {
    (void)n_in; (void)out_size; (void)ws_size;
    const float* ce    = (const float*)d_in[0];
    const float* x     = (const float*)d_in[1];
    const int*   eidx  = (const int*)d_in[2];
    const int*   batch = (const int*)d_in[3];
    const float* W1    = (const float*)d_in[4];
    const float* as1   = (const float*)d_in[5];
    const float* ad1   = (const float*)d_in[6];
    const float* b1    = (const float*)d_in[7];
    const float* W2    = (const float*)d_in[8];
    const float* as2   = (const float*)d_in[9];
    const float* ad2   = (const float*)d_in[10];
    const float* b2    = (const float*)d_in[11];
    const float* gam   = (const float*)d_in[12];
    const float* bet   = (const float*)d_in[13];
    const float* Wc    = (const float*)d_in[14];
    const float* bc    = (const float*)d_in[15];

    const int B  = in_sizes[0] / DIN;
    const int N  = in_sizes[1] / DIN;
    const int E  = in_sizes[2] / 2;
    const int ET = E + N;                    // edges + self loops
    const int* src = eidx;
    const int* dst = eidx + E;

    // -------- workspace layout (256B aligned) --------
    char* wsb = (char*)d_ws;
    size_t o = 0;
#define WS_ALLOC(ty, name, bytes) ty name = (ty)(wsb + o); o += (((size_t)(bytes)) + 255) & ~(size_t)255;
    WS_ALLOC(_Float16*, h0h, (size_t)N * DIN * 2)   // gated feats f16; reused as A2 (N*DH*2)
    WS_ALLOC(_Float16*, Wt,  (size_t)DIN * DH * 2)  // transposed weights (reused W1/W2)
    WS_ALLOC(float*,    h1,  (size_t)N * DH * 4)    // GEMM output (h1, then h2)
    WS_ALLOC(float*,    agg, (size_t)N * DH * 4)    // aggregation output (out1, then out2)
    WS_ALLOC(float*,    als, (size_t)N * 4)
    WS_ALLOC(float*,    ald, (size_t)N * 4)
    WS_ALLOC(int*,      mx,  (size_t)N * 4)
    WS_ALLOC(float*,    den, (size_t)N * 4)
    WS_ALLOC(float*,    elog,(size_t)ET * 4)
    WS_ALLOC(float*,    wgt, (size_t)ET * 4)
    WS_ALLOC(float*,    stat,(size_t)2 * DH * 4)
    WS_ALLOC(float*,    pool,(size_t)B * DH * 4)
    WS_ALLOC(float*,    cnts,(size_t)B * 4)
#undef WS_ALLOC

    const dim3 blk(256);
    const int gWaveN = (N * 32 + 255) / 256;     // one wave per node
    const int gNode  = (N + 255) / 256;
    const int gEdge  = (ET + 255) / 256;
    const int gNDH   = (N * DH + 255) / 256;
    const dim3 gGemm((((N + 15) / 16) + 7) / 8, DH / 64);

    // ---- gating + GEMM1 ----
    qa_gate_kernel<<<gWaveN, blk, 0, stream>>>(ce, x, batch, h0h, N);
    qa_wt_kernel<<<(DIN * DH + 255) / 256, blk, 0, stream>>>(W1, Wt, DIN, DH);
    qa_wmma_gemm_kernel<<<gGemm, blk, 0, stream>>>(h0h, Wt, h1, N, DIN, DH);

    // ---- GAT layer 1 attention + aggregation ----
    qa_alpha_kernel<<<gWaveN, blk, 0, stream>>>(h1, as1, ad1, als, ald, N);
    qa_seg_init_kernel<<<gNode, blk, 0, stream>>>(mx, den, N);
    qa_edge_max_kernel<<<gEdge, blk, 0, stream>>>(src, dst, als, ald, elog, mx, E, N);
    qa_edge_exp_kernel<<<gEdge, blk, 0, stream>>>(src, dst, elog, mx, wgt, den, E, N);
    qa_out_bias_kernel<<<gNDH, blk, 0, stream>>>(agg, b1, N * DH);
    qa_edge_agg_kernel<<<ET, blk, 0, stream>>>(src, dst, h1, wgt, den, agg, E, N);

    // ---- batchnorm + relu -> f16 ----
    qa_bn_stats_kernel<<<DH, blk, 0, stream>>>(agg, stat, N);
    qa_bn_apply_kernel<<<gNDH, blk, 0, stream>>>(agg, stat, gam, bet, h0h, N * DH);

    // ---- GEMM2 ----
    qa_wt_kernel<<<(DH * DH + 255) / 256, blk, 0, stream>>>(W2, Wt, DH, DH);
    qa_wmma_gemm_kernel<<<gGemm, blk, 0, stream>>>(h0h, Wt, h1, N, DH, DH);

    // ---- GAT layer 2 attention + aggregation ----
    qa_alpha_kernel<<<gWaveN, blk, 0, stream>>>(h1, as2, ad2, als, ald, N);
    qa_seg_init_kernel<<<gNode, blk, 0, stream>>>(mx, den, N);
    qa_edge_max_kernel<<<gEdge, blk, 0, stream>>>(src, dst, als, ald, elog, mx, E, N);
    qa_edge_exp_kernel<<<gEdge, blk, 0, stream>>>(src, dst, elog, mx, wgt, den, E, N);
    qa_out_bias_kernel<<<gNDH, blk, 0, stream>>>(agg, b2, N * DH);
    qa_edge_agg_kernel<<<ET, blk, 0, stream>>>(src, dst, h1, wgt, den, agg, E, N);

    // ---- relu + mean pool + final linear ----
    qa_zero_kernel<<<(B * DH + 255) / 256, blk, 0, stream>>>(pool, B * DH);
    qa_zero_kernel<<<(B + 255) / 256, blk, 0, stream>>>(cnts, B);
    qa_counts_kernel<<<gNode, blk, 0, stream>>>(batch, cnts, N);
    qa_pool_kernel<<<gNDH, blk, 0, stream>>>(agg, batch, pool, N * DH);
    qa_final_kernel<<<B, blk, 0, stream>>>(pool, cnts, ce, Wc, bc, (float*)d_out);
}
